// QuantIPA_44555990729116
// MI455X (gfx1250) — matrix operand order repaired and checked by reading, not compiled, not run on hardware
//
#include <hip/hip_runtime.h>
#include <hip/hip_bf16.h>
#include <math.h>

// ---------------- types / helpers ----------------
typedef __attribute__((ext_vector_type(16))) __bf16 v16bf;
typedef __attribute__((ext_vector_type(8)))  float  v8f;
typedef unsigned short u16;
typedef unsigned long long u64t;

constexpr int NN  = 512;
constexpr int CSn = 384;
constexpr int CZn = 128;
constexpr int HHn = 12;
constexpr int PQn = 4;
constexpr int PVn = 8;
constexpr int CZ4n = 32;

__device__ __forceinline__ u16 f2bf(float f) {
  unsigned u = __float_as_uint(f);
  u += 0x7FFFu + ((u >> 16) & 1u);
  return (u16)(u >> 16);
}
__device__ __forceinline__ u64t pack4bf(float a, float b, float c, float d) {
  return (u64t)f2bf(a) | ((u64t)f2bf(b) << 16) | ((u64t)f2bf(c) << 32) | ((u64t)f2bf(d) << 48);
}
__device__ __forceinline__ v8f wmma_bf16(v16bf a, v16bf b, v8f c) {
  return __builtin_amdgcn_wmma_f32_16x16x32_bf16(false, a, false, b, (short)0, c, false, false);
}

// ---- fragment-layout index helpers (CDNA5 ISA 7.12.2) ----
// A 16x32 bf16: (lane,e) -> m = lane&15, k = e + 16*((lane>>4)&1) (contiguous k used here, not ISA table order)
// B 32x16 bf16: (lane,e) -> k = e + 16*((lane>>4)&1), n = lane&15 (contiguous k used here, not ISA table order)
// C 16x16 f32 : (lane,r) -> m = r + 8*((lane>>4)&1), n = lane&15
__device__ __forceinline__ int aswz(int m, int k) {          // (m,k) -> lane*16+e
  int lane = m + (((k >> 4) & 1) << 4);
  int e = k & 15;
  return lane * 16 + e;
}
__device__ __forceinline__ int bswz(int k, int n) {          // (k,n) -> lane*16+e
  int lane = n + (((k >> 4) & 1) << 4);
  return lane * 16 + (k & 15);
}
__device__ __forceinline__ long bbias_idx(int h, int i, int j) { // C-frag order
  int it = i >> 4, m = i & 15, jt = j >> 4;
  int ln = (j & 15) + ((m >> 3) << 4);
  return (((long)h * 32 + it) * 32 + jt) * 256 + ln * 8 + (m & 7);
}
__device__ __forceinline__ long pzB_idx(int i, int j, int c) {   // per-i B-frag
  int ks = j >> 5;
  return (((long)i * 16 + ks) * 2 + (c >> 4)) * 512 + bswz(j & 31, c & 15);
}
__device__ __forceinline__ long aA_idx(int h, int i, int j) {    // per-i A-frag (M=head)
  int ks = j >> 5;
  return ((long)i * 16 + ks) * 512 + aswz(h, j & 31);
}
__device__ __forceinline__ long feat_idx(int i, int col) {       // feats A-frag
  int it = i >> 4, ks = col >> 5;
  return (((long)it * 30 + ks) * 32) * 16 + aswz(i & 15, col & 31);
}

// ---------------- K0: pack weights (pre-swizzled), zero pad regions ----------------
__global__ void ipa_prep(const float* w_b, const float* w_dz, const float* w_out,
                         const float* head_weights, const float* item3,
                         u16* WzB, u16* WoutB, float* hw,
                         u16* qA, u16* kB, u16* vpB, u16* aA) {
  int t = blockIdx.x * blockDim.x + threadIdx.x;
  long stride = (long)gridDim.x * blockDim.x;
  // WzB: [ks 0..3][nt 0..2][lane][e]
  for (long idx = t; idx < 4 * 3 * 512; idx += stride) {
    int ks = (int)(idx / (3 * 512));
    int rem = (int)(idx % (3 * 512));
    int nt = rem / 512;
    int l = (rem >> 4) & 31;
    int e = rem & 15;
    int k = ks * 32 + (((l >> 4) & 1) << 4) + e;
    int n = nt * 16 + (l & 15);
    float v = (n < HHn) ? w_b[k * HHn + n]
            : (n < HHn + CZ4n) ? w_dz[k * CZ4n + (n - HHn)] : 0.f;
    WzB[idx] = f2bf(v);
  }
  // WoutB: [ks 0..29][nt 0..23][lane][e]
  for (long idx = t; idx < (long)30 * 24 * 512; idx += stride) {
    int ks = (int)(idx / (24 * 512));
    int rem = (int)(idx % (24 * 512));
    int nt = rem / 512;
    int l = (rem >> 4) & 31;
    int e = rem & 15;
    int k = ks * 32 + (((l >> 4) & 1) << 4) + e;
    int n = nt * 16 + (l & 15);
    WoutB[idx] = f2bf(w_out[(long)k * CSn + n]);
  }
  // zero-fill padded operand buffers with 32-bit stores
  for (long idx = t; idx < (long)HHn * 32 * 256; idx += stride) {
    ((unsigned*)qA)[idx] = 0; ((unsigned*)kB)[idx] = 0;
  }
  for (long idx = t; idx < (long)HHn * 16 * 512; idx += stride) ((unsigned*)vpB)[idx] = 0;
  for (long idx = t; idx < (long)NN * 16 * 256; idx += stride) ((unsigned*)aA)[idx] = 0;
  if (blockIdx.x == 0 && threadIdx.x < HHn) {
    float x = head_weights[threadIdx.x];
    float sp = (x > 20.f) ? x : log1pf(expf(x));
    hw[threadIdx.x] = sp * sqrtf(1.f / (PQn * 9.f / 2.f)) * item3[0];
  }
}

// ---------------- K1: per-residue projections + rigid transforms ----------------
__global__ void ipa_proj(const float* s, const float* mask, const float* rot, const float* trans,
                         const float* w_q, const float* b_q, const float* w_kv, const float* b_kv,
                         const float* w_qp, const float* b_qp, const float* w_kvp, const float* b_kvp,
                         u16* qA, u16* kB, u16* vB, u16* vpB, float* qp_f, float* kp_f) {
  int n = blockIdx.x;
  int t = threadIdx.x; // 256
  __shared__ alignas(16) float srow[CSn];
  __shared__ float praw[1152];
  float mk = mask[n];
  for (int c = t; c < CSn / 4; c += 256)
    reinterpret_cast<float4*>(srow)[c] = reinterpret_cast<const float4*>(s + (long)n * CSn)[c];
  __syncthreads();
  for (int f = t; f < 1152; f += 256) {
    const float *w, *b; int col, st;
    if (f < 192)      { w = w_q;   b = b_q;   col = f;       st = 192; }
    else if (f < 576) { w = w_kv;  b = b_kv;  col = f - 192; st = 384; }
    else if (f < 720) { w = w_qp;  b = b_qp;  col = f - 576; st = 144; }
    else              { w = w_kvp; b = b_kvp; col = f - 720; st = 432; }
    float acc = b[col];
    for (int k = 0; k < CSn; ++k) acc = fmaf(srow[k], w[k * st + col], acc);
    praw[f] = acc * mk;
  }
  __syncthreads();
  int it = n >> 4, m = n & 15;       // A-side coordinates for q
  int nc = n & 15;                   // B-side (logits) column for k
  int ks = n >> 5, kb = n & 31;      // B-side (o-GEMM) K coords for v / v_pts
  for (int f = t; f < 576; f += 256) {
    u16 bv = f2bf(praw[f]);
    if (f < 192) {                       // q -> A-frag order [h][it][lane*16+e]
      int h = f >> 4, c = f & 15;
      qA[((long)(h * 32 + it) * 32) * 16 + aswz(m, c)] = bv;
    } else {
      int g = f - 192, h = g / 32, c2 = g % 32;
      if (c2 < 16)                       // k -> B-frag (K = channel) [h][jt][...]
        kB[((long)(h * 32 + it) * 32 + nc) * 16 + c2] = bv;
      else                               // v -> B-frag (K = j) [h][ks][...]
        vB[((long)h * 16 + ks) * 512 + bswz(kb, c2 - 16)] = bv;
    }
  }
  float R0=rot[n*9+0],R1=rot[n*9+1],R2=rot[n*9+2],R3=rot[n*9+3],R4=rot[n*9+4];
  float R5=rot[n*9+5],R6=rot[n*9+6],R7=rot[n*9+7],R8=rot[n*9+8];
  float T0=trans[n*3+0],T1=trans[n*3+1],T2=trans[n*3+2];
  for (int hp = t; hp < 48; hp += 256) { // q_pts: 144 feats = [3][48]
    float lx = praw[576 + hp], ly = praw[576 + 48 + hp], lz = praw[576 + 96 + hp];
    float gx = R0*lx + R1*ly + R2*lz + T0;
    float gy = R3*lx + R4*ly + R5*lz + T1;
    float gz = R6*lx + R7*ly + R8*lz + T2;
    int h = hp >> 2, p = hp & 3;
    float* d = qp_f + ((long)h * NN + n) * 12 + p * 3;
    d[0] = gx; d[1] = gy; d[2] = gz;
  }
  for (int hp = t; hp < 144; hp += 256) { // kv_pts: 432 feats = [3][144]
    float lx = praw[720 + hp], ly = praw[720 + 144 + hp], lz = praw[720 + 288 + hp];
    float gx = R0*lx + R1*ly + R2*lz + T0;
    float gy = R3*lx + R4*ly + R5*lz + T1;
    float gz = R6*lx + R7*ly + R8*lz + T2;
    int h = hp / 12, pt = hp % 12;
    if (pt < PQn) {
      float* d = kp_f + ((long)h * NN + n) * 12 + pt * 3;
      d[0] = gx; d[1] = gy; d[2] = gz;
    } else {
      int col = (pt - PQn) * 3;
      float g3v[3] = {gx, gy, gz};
      for (int d2 = 0; d2 < 3; ++d2) {
        int c = col + d2;
        vpB[(((long)h * 2 + (c >> 4)) * 16 + ks) * 512 + bswz(kb, c & 15)] = f2bf(g3v[d2]);
      }
    }
  }
}

// ---------------- K2: fused z pass -> bbias (C-frag) + pair_z (B-frag) ----------------
__global__ void ipa_zproj(const float* z, const float* mask, const float* b_b, const float* b_dz,
                          const u16* WzB, float* bbias, u16* pzB) {
  int m0 = blockIdx.x * 16;     // flattened ij base; i constant within block
  int lane = threadIdx.x;       // 32
  __shared__ alignas(32) u16 zt[4 * 512];  // A-swizzled z tile
  // float4 loads + packed 8B bf16x4 LDS stores (4 consecutive K -> 4 consecutive e)
  const float4* z4 = reinterpret_cast<const float4*>(z + (long)m0 * CZn);
  for (int idx = lane; idx < 16 * (CZn / 4); idx += 32) {
    int r = idx >> 5;             // 32 float4 per row
    int c0 = (idx & 31) << 2;
    float4 v = z4[idx];
    *reinterpret_cast<u64t*>(&zt[(c0 >> 5) * 512 + aswz(r, c0 & 31)]) =
        pack4bf(v.x, v.y, v.z, v.w);
  }
  __syncthreads();
  v8f acc0 = {}, acc1 = {}, acc2 = {};
  for (int ksi = 0; ksi < 4; ++ksi) {
    v16bf A  = *reinterpret_cast<const v16bf*>(zt + ksi * 512 + lane * 16);
    v16bf B0 = *reinterpret_cast<const v16bf*>(WzB + ((ksi * 3 + 0) * 512 + lane * 16));
    v16bf B1 = *reinterpret_cast<const v16bf*>(WzB + ((ksi * 3 + 1) * 512 + lane * 16));
    v16bf B2 = *reinterpret_cast<const v16bf*>(WzB + ((ksi * 3 + 2) * 512 + lane * 16));
    acc0 = wmma_bf16(A, B0, acc0);
    acc1 = wmma_bf16(A, B1, acc1);
    acc2 = wmma_bf16(A, B2, acc2);
  }
  int i = m0 >> 9, j0z = m0 & 511;
  float mi = mask[i];
  int nb = lane & 15;
#pragma unroll
  for (int r = 0; r < 8; ++r) {
    int m = r + ((lane & 16) ? 8 : 0);
    int j = j0z + m;
    float em = mi * mask[j];
    { int nn = nb; float v = acc0[r];
      if (nn < HHn) bbias[bbias_idx(nn, i, j)] = (v + b_b[nn]) * em;
      else pzB[pzB_idx(i, j, nn - HHn)] = f2bf((v + b_dz[nn - HHn]) * em); }
    { int nn = 16 + nb; float v = acc1[r];
      pzB[pzB_idx(i, j, nn - HHn)] = f2bf((v + b_dz[nn - HHn]) * em); }
    { int nn = 32 + nb; float v = acc2[r];
      if (nn < HHn + CZ4n) pzB[pzB_idx(i, j, nn - HHn)] = f2bf((v + b_dz[nn - HHn]) * em); }
  }
}

// ---------------- K3: fused attention: logits + softmax + o + o_pt ----------------
__global__ void ipa_attn(const u16* qA, const u16* kB, const u16* vB, const u16* vpB,
                         const float* qp_f, const float* kp_f, const float* bbias,
                         const float* mask, const float* rot, const float* trans,
                         const float* hw, const float* item1, const float* item2,
                         u16* aA, u16* featsA) {
  int it = blockIdx.x, i0 = it * 16;
  int h  = blockIdx.y;
  int t = threadIdx.x;            // 128 = 4 waves
  int lane = t & 31, wave = t >> 5;

  __shared__ alignas(32) float L[16][NN];     // 32 KB logits
  __shared__ alignas(32) u16 aSw[16 * 512];   // 16 KB softmaxed a, A-swizzled
  __shared__ float qp[16 * 12];
  __shared__ float ml[16];
  __shared__ float red[16 * 8];
  __shared__ float optl[16 * 24];

  float scale_qk = 0.25f * item1[0];
  float it2 = item2[0];
  float hwh = hw[h];

  for (int idx = t; idx < 16 * 12; idx += 128) qp[idx] = qp_f[((long)h * NN + i0) * 12 + idx];
  if (t < 16) ml[t] = mask[i0 + t];
  __syncthreads();

  // A-frag: q tile (K padded with prep's zeros)
  v16bf Aq = *reinterpret_cast<const v16bf*>(qA + ((long)(h * 32 + it) * 32 + lane) * 16);

  // ---- logits ----
  for (int jt = wave; jt < 32; jt += 4) {
    v16bf Bk = *reinterpret_cast<const v16bf*>(kB + ((long)(h * 32 + jt) * 32 + lane) * 16);
    v8f Cf = {};
    Cf = wmma_bf16(Aq, Bk, Cf);
    v8f bb = *reinterpret_cast<const v8f*>(bbias + (((long)h * 32 + it) * 32 + jt) * 256 + lane * 8);
    int nn = lane & 15;
    int j = jt * 16 + nn;
    const float4* kp4 = reinterpret_cast<const float4*>(kp_f + ((long)h * NN + j) * 12);
    float4 ka = kp4[0], kbv = kp4[1], kc = kp4[2];
    float mj = mask[j];
#pragma unroll
    for (int r = 0; r < 8; ++r) {
      int m = r + ((lane & 16) ? 8 : 0);
      const float* q0 = qp + m * 12;
      float dx, dy, dz, pa;
      dx = q0[0]-ka.x;  dy = q0[1]-ka.y;  dz = q0[2]-ka.z;  pa  = dx*dx+dy*dy+dz*dz;
      dx = q0[3]-ka.w;  dy = q0[4]-kbv.x; dz = q0[5]-kbv.y; pa += dx*dx+dy*dy+dz*dz;
      dx = q0[6]-kbv.z; dy = q0[7]-kbv.w; dz = q0[8]-kc.x;  pa += dx*dx+dy*dy+dz*dz;
      dx = q0[9]-kc.y;  dy = q0[10]-kc.z; dz = q0[11]-kc.w; pa += dx*dx+dy*dy+dz*dz;
      L[m][j] = Cf[r] * scale_qk + it2 * bb[r] - 0.5f * hwh * pa
              + 100000.f * (ml[m] * mj - 1.f);
    }
  }
  __syncthreads();

  // ---- softmax over j (float4 passes; packed bf16x4 stores A-swizzled) ----
  {
    int row = t & 15, ch = t >> 4; // 8 chunks of 16 float4
    float4* Lr = reinterpret_cast<float4*>(&L[row][0]);
    float mx = -3.4e38f;
    for (int q4 = ch * 16; q4 < ch * 16 + 16; ++q4) {
      float4 v = Lr[q4];
      mx = fmaxf(fmaxf(fmaxf(mx, v.x), v.y), fmaxf(v.z, v.w));
    }
    red[row * 8 + ch] = mx;
    __syncthreads();
    float rmx = red[row * 8];
    for (int c = 1; c < 8; ++c) rmx = fmaxf(rmx, red[row * 8 + c]);
    float sum = 0.f;
    for (int q4 = ch * 16; q4 < ch * 16 + 16; ++q4) {
      float4 v = Lr[q4];
      v.x = __expf(v.x - rmx); v.y = __expf(v.y - rmx);
      v.z = __expf(v.z - rmx); v.w = __expf(v.w - rmx);
      sum += (v.x + v.y) + (v.z + v.w);
      Lr[q4] = v;
    }
    __syncthreads();
    red[row * 8 + ch] = sum;
    __syncthreads();
    float rs = 0.f;
    for (int c = 0; c < 8; ++c) rs += red[row * 8 + c];
    float inv = 1.f / rs;
    for (int q4 = ch * 16; q4 < ch * 16 + 16; ++q4) {
      float4 v = Lr[q4];
      int j = q4 << 2;
      u64t p = pack4bf(v.x * inv, v.y * inv, v.z * inv, v.w * inv);
      *reinterpret_cast<u64t*>(&aSw[(j >> 5) * 512 + aswz(row, j & 31)]) = p;
      *reinterpret_cast<u64t*>(&aA[aA_idx(h, i0 + row, j)]) = p;
    }
  }
  __syncthreads();

  // ---- o = A@V (wave0), o_pt = A@Vpts (waves 1,2) ----
  if (wave < 3) {
    const u16* Bbase = (wave == 0) ? (vB + (long)h * 16 * 512)
                                   : (vpB + ((long)h * 2 + (wave - 1)) * 16 * 512);
    v8f acc = {};
    for (int ksi = 0; ksi < 16; ++ksi) {
      v16bf Af = *reinterpret_cast<const v16bf*>(aSw + ksi * 512 + lane * 16);
      v16bf Bf = *reinterpret_cast<const v16bf*>(Bbase + (long)ksi * 512 + lane * 16);
      acc = wmma_bf16(Af, Bf, acc);
    }
#pragma unroll
    for (int r = 0; r < 8; ++r) {
      int m = r + ((lane & 16) ? 8 : 0);
      int nn = lane & 15;
      if (wave == 0)      featsA[feat_idx(i0 + m, h * 16 + nn)] = f2bf(acc[r]);
      else if (wave == 1) optl[m * 24 + nn] = acc[r];
      else if (nn < 8)    optl[m * 24 + 16 + nn] = acc[r];
    }
  }
  __syncthreads();

  // ---- o_pt epilogue: inverse rigid transform + norms ----
  {
    int m = t >> 3, pt = t & 7; // 16x8 = 128
    int i = i0 + m;
    const float* Rp = rot + (long)i * 9;
    const float* Tp = trans + (long)i * 3;
    float gx = optl[m * 24 + pt * 3 + 0] - Tp[0];
    float gy = optl[m * 24 + pt * 3 + 1] - Tp[1];
    float gz = optl[m * 24 + pt * 3 + 2] - Tp[2];
    float lx = Rp[0] * gx + Rp[3] * gy + Rp[6] * gz;  // R^T
    float ly = Rp[1] * gx + Rp[4] * gy + Rp[7] * gz;
    float lz = Rp[2] * gx + Rp[5] * gy + Rp[8] * gz;
    float nr = sqrtf(lx * lx + ly * ly + lz * lz + 1e-8f);
    int hp = h * PVn + pt;
    featsA[feat_idx(i, 192 + hp)] = f2bf(lx);
    featsA[feat_idx(i, 288 + hp)] = f2bf(ly);
    featsA[feat_idx(i, 384 + hp)] = f2bf(lz);
    featsA[feat_idx(i, 480 + hp)] = f2bf(nr);
  }
}

// ---------------- K4: o_pair — per-i GEMM [12x512]@[512x32] ----------------
__global__ void ipa_opair(const u16* aA, const u16* pzB, u16* featsA) {
  int i = blockIdx.x;
  int t = threadIdx.x;            // 64 = 2 waves
  int lane = t & 31, nt = t >> 5; // wave == n-tile
  v8f acc = {};
  for (int ksi = 0; ksi < 16; ++ksi) {
    v16bf Af = *reinterpret_cast<const v16bf*>(aA + ((long)i * 16 + ksi) * 512 + lane * 16);
    v16bf Bf = *reinterpret_cast<const v16bf*>(pzB + (((long)i * 16 + ksi) * 2 + nt) * 512 + lane * 16);
    acc = wmma_bf16(Af, Bf, acc);
  }
#pragma unroll
  for (int r = 0; r < 8; ++r) {
    int hh = r + ((lane & 16) ? 8 : 0);
    int nn = (lane & 15) + nt * 16;
    if (hh < HHn) featsA[feat_idx(i, 576 + hh * CZ4n + nn)] = f2bf(acc[r]);
  }
}

// ---------------- K5: output projection feats[512x960] @ Wout[960x384] ----------------
__global__ void ipa_out(const u16* featsA, const u16* WoutB, const float* b_out,
                        const float* mask, float* out) {
  int it = blockIdx.x, nt = blockIdx.y;
  int lane = threadIdx.x; // 32
  v8f acc = {};
  for (int ksi = 0; ksi < 30; ++ksi) {
    v16bf Af = *reinterpret_cast<const v16bf*>(featsA + (((long)it * 30 + ksi) * 32 + lane) * 16);
    v16bf Bf = *reinterpret_cast<const v16bf*>(WoutB + (((long)ksi * 24 + nt) * 32 + lane) * 16);
    acc = wmma_bf16(Af, Bf, acc);
  }
#pragma unroll
  for (int r = 0; r < 8; ++r) {
    int m = r + ((lane & 16) ? 8 : 0);
    int nn = nt * 16 + (lane & 15);
    int i = it * 16 + m;
    out[(long)i * CSn + nn] = (acc[r] + b_out[nn]) * mask[i];
  }
}

// ---------------- launch ----------------
extern "C" void kernel_launch(void* const* d_in, const int* in_sizes, int n_in,
                              void* d_out, int out_size, void* d_ws, size_t ws_size,
                              hipStream_t stream) {
  const float* s      = (const float*)d_in[0];
  const float* z      = (const float*)d_in[1];
  const float* mask   = (const float*)d_in[2];
  const float* rot    = (const float*)d_in[3];
  const float* trans  = (const float*)d_in[4];
  const float* w_q    = (const float*)d_in[5];
  const float* b_q    = (const float*)d_in[6];
  const float* w_kv   = (const float*)d_in[7];
  const float* b_kv   = (const float*)d_in[8];
  const float* w_qp   = (const float*)d_in[9];
  const float* b_qp   = (const float*)d_in[10];
  const float* w_kvp  = (const float*)d_in[11];
  const float* b_kvp  = (const float*)d_in[12];
  const float* w_b    = (const float*)d_in[13];
  const float* b_b    = (const float*)d_in[14];
  const float* w_dz   = (const float*)d_in[15];
  const float* b_dz   = (const float*)d_in[16];
  const float* w_out  = (const float*)d_in[17];
  const float* b_out  = (const float*)d_in[18];
  const float* hweights = (const float*)d_in[19];
  const float* item1  = (const float*)d_in[20];
  const float* item2  = (const float*)d_in[21];
  const float* item3  = (const float*)d_in[22];
  float* out = (float*)d_out;

  char* ws = (char*)d_ws;
  size_t off = 0;
  auto alloc = [&](size_t bytes) -> void* {
    void* p = ws + off;
    off = (off + bytes + 255) & ~(size_t)255;
    return p;
  };
  u16*   qAb   = (u16*)alloc((size_t)HHn * 32 * 512 * 2);        // q, A-frag
  u16*   kBb   = (u16*)alloc((size_t)HHn * 32 * 512 * 2);        // k, B-frag (K=channel)
  u16*   vBb   = (u16*)alloc((size_t)HHn * 16 * 512 * 2);        // v, B-frag (K=j)
  u16*   vpBb  = (u16*)alloc((size_t)HHn * 2 * 16 * 512 * 2);    // v_pts, B-frag, padded
  float* qp_f  = (float*)alloc((size_t)HHn * NN * 12 * 4);
  float* kp_f  = (float*)alloc((size_t)HHn * NN * 12 * 4);
  u16*   WzB   = (u16*)alloc((size_t)4 * 3 * 512 * 2);
  u16*   WoutB = (u16*)alloc((size_t)30 * 24 * 512 * 2);
  float* hw    = (float*)alloc(256);
  float* bbias = (float*)alloc((size_t)HHn * NN * NN * 4);       // C-frag order
  u16*   pzB   = (u16*)alloc((size_t)NN * 16 * 2 * 512 * 2);     // B-frag order per i
  u16*   aA    = (u16*)alloc((size_t)NN * 16 * 512 * 2);         // A-frag order per i
  u16*   featsA= (u16*)alloc((size_t)32 * 30 * 512 * 2);         // A-frag order

  ipa_prep<<<2048, 256, 0, stream>>>(w_b, w_dz, w_out, hweights, item3,
                                     WzB, WoutB, hw, qAb, kBb, vpBb, aA);
  ipa_proj<<<NN, 256, 0, stream>>>(s, mask, rot, trans, w_q, b_q, w_kv, b_kv,
                                   w_qp, b_qp, w_kvp, b_kvp,
                                   qAb, kBb, vBb, vpBb, qp_f, kp_f);
  ipa_zproj<<<(NN * NN) / 16, 32, 0, stream>>>(z, mask, b_b, b_dz, WzB, bbias, pzB);
  dim3 g3(NN / 16, HHn);
  ipa_attn<<<g3, 128, 0, stream>>>(qAb, kBb, vBb, vpBb, qp_f, kp_f, bbias,
                                   mask, rot, trans, hw, item1, item2, aA, featsA);
  ipa_opair<<<NN, 64, 0, stream>>>(aA, pzB, featsA);
  dim3 g5(NN / 16, CSn / 16);
  ipa_out<<<g5, 32, 0, stream>>>(featsA, WoutB, b_out, mask, out);
}
